// DenseMLP_14912126451779
// MI455X (gfx1250) — matrix operand order repaired and checked by reading, not compiled, not run on hardware
//
#include <hip/hip_runtime.h>
#include <hip/hip_bf16.h>

typedef __attribute__((ext_vector_type(16))) _Float16 v16h;
typedef __attribute__((ext_vector_type(8)))  _Float16 v8h;
typedef __attribute__((ext_vector_type(8)))  float    v8f;

#define B_ROWS   65536
#define IN_DIM   784
#define KPAD     800      // 784 padded to 25 chunks of 32
#define H_DIM    64
#define OUT_DIM  10
#define NL       8
#define RS       72       // LDS row stride in f16 (conflict-free padding)
#define ROWS_PER_BLOCK 32

// workspace layout (bytes)
#define WS_W0H_OFF    0                               // 64 x 800 f16
#define WS_W0H_BYTES  (64 * KPAD * 2)
#define WS_WPH_OFF    (WS_W0H_OFF + WS_W0H_BYTES)     // 49 slots of 64x64 f16: slot (i-1)*7+j
#define WS_WPH_BYTES  (49 * 64 * 64 * 2)
#define WS_WFC_OFF    (WS_WPH_OFF + WS_WPH_BYTES)     // 49 slots of 64x64 f16: slot i*7+k
#define WS_WFC_BYTES  (49 * 64 * 64 * 2)
#define WS_WOUT_OFF   (WS_WFC_OFF + WS_WFC_BYTES)     // 16 x 64 f16 (N padded 10->16)
#define WS_WOUT_BYTES (16 * 64 * 2)

// ---------------- prep kernels (tiny, recomputed every launch) ----------------

// W0 (64 x 784 f32) -> W0h (64 x 800 f16, zero padded K)
__global__ __launch_bounds__(256)
void prep_w0(const float* __restrict__ W0, _Float16* __restrict__ W0h) {
  int e = blockIdx.x * 256 + threadIdx.x;        // 0 .. 64*800-1 (exact)
  int n = e / KPAD, k = e % KPAD;
  float v = (k < IN_DIM) ? W0[n * IN_DIM + k] : 0.0f;
  W0h[e] = (_Float16)v;
}

// WpH[(i-1)*7+j] = Wh[i-1] @ Wp[j*7+(i-1)]   (fuses past-matmul with Wh-matmul)
__global__ __launch_bounds__(256)
void prep_wph(const float* __restrict__ Wh, const float* __restrict__ Wp,
              _Float16* __restrict__ WpH) {
  int slot = blockIdx.x >> 4;                    // 0..48
  int i1 = slot / 7;                             // i-1 in 0..6
  int j  = slot % 7;
  if (j > i1) return;                            // valid: j < i  <=>  j <= i1
  int e = ((blockIdx.x & 15) << 8) + threadIdx.x; // 0..4095
  int n = e >> 6, q = e & 63;
  const float* A  = Wh + i1 * 4096;              // (n,p)
  const float* Bm = Wp + (j * 7 + i1) * 4096;    // (p,q)
  float s = 0.0f;
  #pragma unroll
  for (int p = 0; p < 64; ++p) s += A[n * 64 + p] * Bm[p * 64 + q];
  WpH[slot * 4096 + e] = (_Float16)s;
}

// WfC[i*7+k] = sum_{j=i+1..7} Wf[k*7+(j-1)]   (collapses the future double-sum)
__global__ __launch_bounds__(256)
void prep_wfc(const float* __restrict__ Wf, _Float16* __restrict__ WfC) {
  int slot = blockIdx.x >> 4;                    // 0..41
  int i = slot / 7 + 1;                          // 1..6
  int k = slot % 7;
  if (k > i) return;                             // valid: k <= i
  int e = ((blockIdx.x & 15) << 8) + threadIdx.x;
  float s = 0.0f;
  for (int j = i + 1; j <= 7; ++j) s += Wf[(k * 7 + (j - 1)) * 4096 + e];
  WfC[(i * 7 + k) * 4096 + e] = (_Float16)s;
}

// Wout (10 x 64 f32) -> WoutH (16 x 64 f16, rows 10..15 zero)
__global__ __launch_bounds__(256)
void prep_wout(const float* __restrict__ Wout, _Float16* __restrict__ WoutH) {
  int e = blockIdx.x * 256 + threadIdx.x;        // 0..1023 (grid 4, exact)
  int n = e >> 6, q = e & 63;
  float v = (n < OUT_DIM) ? Wout[n * 64 + q] : 0.0f;
  WoutH[e] = (_Float16)v;
}

// ---------------- main fused kernel ----------------

__device__ __forceinline__ v16h lds_A(const _Float16* p) {
  // CDNA5 f16 A-operand 16x32: lanes 0-15 hold K 0..7 / 16..23, lanes 16-31 K 8..15 / 24..31
  v8h lo = *(const v8h*)p;
  v8h hi = *(const v8h*)(p + 16);
  return __builtin_shufflevector(lo, hi, 0,1,2,3,4,5,6,7,8,9,10,11,12,13,14,15);
}

__device__ __forceinline__ v8f wmma_f16(v16h a, v16h b, v8f c) {
  return __builtin_amdgcn_wmma_f32_16x16x32_f16(false, a, false, b, (short)0, c, false, false);
}

__device__ __forceinline__ float silu(float t) {
  return t / (1.0f + __expf(-t));
}

__global__ __launch_bounds__(256)
void densemlp_main(const float* __restrict__ x,
                   const float* __restrict__ b0,
                   const float* __restrict__ bh,
                   const float* __restrict__ bout,
                   const _Float16* __restrict__ W0h,
                   const _Float16* __restrict__ WpH,
                   const _Float16* __restrict__ WfC,
                   const _Float16* __restrict__ WoutH,
                   float* __restrict__ out) {
  __shared__ __align__(16) _Float16 act[NL][ROWS_PER_BLOCK][RS];
  __shared__ float lgbuf[ROWS_PER_BLOCK][OUT_DIM];

  const int tid    = threadIdx.x;
  const int wave   = tid >> 5;
  const int lane   = tid & 31;
  const int laneLo = lane & 15;
  const int laneHi = lane >> 4;
  const int rgrp   = wave >> 2;             // 0..1  (row group of 16)
  const int ntile  = wave & 3;              // 0..3  (16-col tile)
  const int rowA   = rgrp * 16 + laneLo;            // A-operand row (block-local)
  const int colC   = ntile * 16 + laneLo;           // C/D column
  const int rowC0  = rgrp * 16 + laneHi * 8;        // C/D row base (+v)
  const long long row0 = (long long)blockIdx.x * ROWS_PER_BLOCK;

  // ---------- layer 0: h0 = silu(x @ W0.T + b0), K = 784 (25 chunks of 32) ----------
  {
    v8f acc = {};
    const float* xr = x + (row0 + rowA) * IN_DIM;
    const _Float16* w0r = W0h + (size_t)colC * KPAD;
    #pragma unroll
    for (int c = 0; c < 25; ++c) {
      int k0 = c * 32 + laneHi * 8;
      float4 a0 = *(const float4*)(xr + k0);
      float4 a1 = *(const float4*)(xr + k0 + 4);
      float4 g0, g1;
      if (c < 24) {
        g0 = *(const float4*)(xr + k0 + 16);
        g1 = *(const float4*)(xr + k0 + 20);
      } else {  // tail: K=784..799 is zero padding
        g0 = make_float4(0.f, 0.f, 0.f, 0.f);
        g1 = make_float4(0.f, 0.f, 0.f, 0.f);
      }
      v16h a;
      a[0]=(_Float16)a0.x; a[1]=(_Float16)a0.y; a[2]=(_Float16)a0.z; a[3]=(_Float16)a0.w;
      a[4]=(_Float16)a1.x; a[5]=(_Float16)a1.y; a[6]=(_Float16)a1.z; a[7]=(_Float16)a1.w;
      a[8]=(_Float16)g0.x; a[9]=(_Float16)g0.y; a[10]=(_Float16)g0.z; a[11]=(_Float16)g0.w;
      a[12]=(_Float16)g1.x; a[13]=(_Float16)g1.y; a[14]=(_Float16)g1.z; a[15]=(_Float16)g1.w;
      v16h b = lds_A(w0r + c * 32 + laneHi * 8);
      acc = wmma_f16(a, b, acc);
    }
    float bs = b0[colC];
    #pragma unroll
    for (int v = 0; v < 8; ++v) {
      float t = silu(acc[v] + bs);
      act[0][rowC0 + v][colC] = (_Float16)t;
    }
  }
  __syncthreads();

  // ---------- layers 1..7 ----------
  #pragma unroll
  for (int i = 1; i < NL; ++i) {
    // h_pre = silu( sum_{j<i} out_j @ WpH[i][j].T + bh[i-1] )
    v8f accP = {};
    #pragma unroll
    for (int j = 0; j < i; ++j) {
      const _Float16* Wt = WpH + ((i - 1) * 7 + j) * 4096;
      #pragma unroll
      for (int c = 0; c < 2; ++c) {
        v16h a = lds_A(&act[j][rowA][c * 32 + laneHi * 8]);
        v16h b = lds_A(Wt + colC * 64 + c * 32 + laneHi * 8);
        accP = wmma_f16(a, b, accP);
      }
    }
    float bs = bh[(i - 1) * H_DIM + colC];
    float hpre[8];
    #pragma unroll
    for (int v = 0; v < 8; ++v) {
      float t = silu(accP[v] + bs);
      hpre[v] = t;
      act[i][rowC0 + v][colC] = (_Float16)t;   // pre-add version visible to future sum
    }
    __syncthreads();

    if (i < NL - 1) {
      // F = sum_{k<=i} out_k @ WfC[i][k].T   (uses pre-add act[i], post-add act[k<i])
      v8f accF = {};
      #pragma unroll
      for (int k = 0; k <= i; ++k) {
        const _Float16* Wt = WfC + (i * 7 + k) * 4096;
        #pragma unroll
        for (int c = 0; c < 2; ++c) {
          v16h a = lds_A(&act[k][rowA][c * 32 + laneHi * 8]);
          v16h b = lds_A(Wt + colC * 64 + c * 32 + laneHi * 8);
          accF = wmma_f16(a, b, accF);
        }
      }
      __syncthreads();  // all reads of pre-add act[i] done before overwrite
      #pragma unroll
      for (int v = 0; v < 8; ++v)
        act[i][rowC0 + v][colC] = (_Float16)(hpre[v] + accF[v]);
      __syncthreads();
    }
  }

  // ---------- output head on WMMA: logits = act7 @ WoutH.T (N padded to 16) ----------
  // ntile==0 is wave-uniform, so EXEC is all-ones inside (required for WMMA).
  if (ntile == 0) {
    v8f accO = {};
    #pragma unroll
    for (int c = 0; c < 2; ++c) {
      v16h a = lds_A(&act[NL - 1][rowA][c * 32 + laneHi * 8]);
      v16h b = lds_A(WoutH + laneLo * 64 + c * 32 + laneHi * 8);
      accO = wmma_f16(a, b, accO);
    }
    if (laneLo < OUT_DIM) {
      #pragma unroll
      for (int v = 0; v < 8; ++v)
        lgbuf[rowC0 + v][laneLo] = accO[v];
    }
  }
  __syncthreads();

  // ---------- log_softmax over 10 classes, one thread per row ----------
  if (tid < ROWS_PER_BLOCK) {
    float lg[OUT_DIM];
    #pragma unroll
    for (int o = 0; o < OUT_DIM; ++o) lg[o] = lgbuf[tid][o] + bout[o];
    float mx = lg[0];
    #pragma unroll
    for (int o = 1; o < OUT_DIM; ++o) mx = fmaxf(mx, lg[o]);
    float se = 0.0f;
    #pragma unroll
    for (int o = 0; o < OUT_DIM; ++o) se += __expf(lg[o] - mx);
    float lse = __logf(se) + mx;
    float* orow = out + (row0 + tid) * OUT_DIM;
    #pragma unroll
    for (int o = 0; o < OUT_DIM; ++o) orow[o] = lg[o] - lse;
  }
}

// ---------------- launch ----------------

extern "C" void kernel_launch(void* const* d_in, const int* in_sizes, int n_in,
                              void* d_out, int out_size, void* d_ws, size_t ws_size,
                              hipStream_t stream) {
  (void)in_sizes; (void)n_in; (void)out_size; (void)ws_size;
  const float* x    = (const float*)d_in[0];
  const float* W0   = (const float*)d_in[1];
  const float* b0   = (const float*)d_in[2];
  const float* Wh   = (const float*)d_in[3];
  const float* bh   = (const float*)d_in[4];
  const float* Wp   = (const float*)d_in[5];
  const float* Wf   = (const float*)d_in[6];
  const float* Wout = (const float*)d_in[7];
  const float* bout = (const float*)d_in[8];
  float* out = (float*)d_out;

  char* ws = (char*)d_ws;
  _Float16* W0h   = (_Float16*)(ws + WS_W0H_OFF);
  _Float16* WpH   = (_Float16*)(ws + WS_WPH_OFF);
  _Float16* WfC   = (_Float16*)(ws + WS_WFC_OFF);
  _Float16* WoutH = (_Float16*)(ws + WS_WOUT_OFF);

  prep_w0  <<<(64 * KPAD) / 256, 256, 0, stream>>>(W0, W0h);
  prep_wph <<<49 * 16,           256, 0, stream>>>(Wh, Wp, WpH);
  prep_wfc <<<42 * 16,           256, 0, stream>>>(Wf, WfC);
  prep_wout<<<4,                 256, 0, stream>>>(Wout, WoutH);

  densemlp_main<<<B_ROWS / ROWS_PER_BLOCK, 256, 0, stream>>>(
      x, b0, bh, bout, W0h, WpH, WfC, WoutH, out);
}
